// EnhancedLegalRGCN_57750130262357
// MI455X (gfx1250) — matrix operand order repaired and checked by reading, not compiled, not run on hardware
//
#include <hip/hip_runtime.h>
#include <hip/hip_bf16.h>

// ---------------------------------------------------------------------------
// EnhancedLegalRGCN on MI455X (gfx1250, wave32, WMMA)
//
// Roofline: ~49 GFLOP of GEMM vs ~3.5-4 GB of gather/scatter traffic
// -> memory/atomic bound (>=160us at 23.3 TB/s). All GEMMs run on
// v_wmma_f32_16x16x32_bf16 with pre-packed, zero-padded bf16 operands so the
// inner loop is pure global_load_b128 + v_wmma (no guards, no mid-loop waits).
// Segment mean uses global_atomic_add_f32, matching jax segment_sum.
// ---------------------------------------------------------------------------

#define N_NODES 50000
#define E_EDGES 1600000
#define N_REL   3

typedef __attribute__((ext_vector_type(16))) __bf16 v16bf;
typedef __attribute__((ext_vector_type(8)))  __bf16 v8bf;
typedef __attribute__((ext_vector_type(8)))  float  v8f;

// ---------------------------------------------------------------------------
// Pack f32 weights [nslab][K][64] into WMMA-B-swizzled bf16:
//   dst[slab][kt][nt][lane][i] = W[slab][kt*32 + (lane>>4)*8 + (i&7) + (i>>3)*16][nt*16 + (lane&15)]
// (zero beyond K). One thread per output element.
__global__ void k_pack_w(const float* __restrict__ W, __bf16* __restrict__ dst,
                         int K, int Kp, int nslab) {
  int ktiles = Kp >> 5;
  size_t total = (size_t)nslab * ktiles * 2048;
  size_t o = (size_t)blockIdx.x * blockDim.x + threadIdx.x;
  if (o >= total) return;
  int i    = (int)(o & 15);
  int lane = (int)((o >> 4) & 31);
  int nt   = (int)((o >> 9) & 3);
  size_t rest = o >> 11;
  int kt   = (int)(rest % ktiles);
  int slab = (int)(rest / ktiles);
  int k = kt * 32 + ((lane >> 4) << 3) + (i & 7) + ((i >> 3) << 4);
  int n = nt * 16 + (lane & 15);
  dst[o] = (k < K) ? (__bf16)W[((size_t)slab * K + k) * 64 + n] : (__bf16)0.0f;
}

// Pack f32 activations [M][K] -> bf16 [M][Kp], zero-padded.
__global__ void k_pack_act(const float* __restrict__ src, __bf16* __restrict__ dst,
                           int K, int Kp, size_t M) {
  size_t o = (size_t)blockIdx.x * blockDim.x + threadIdx.x;
  if (o >= M * (size_t)Kp) return;
  size_t r = o / (size_t)Kp;
  int k = (int)(o - r * (size_t)Kp);
  dst[o] = (k < K) ? (__bf16)src[r * (size_t)K + k] : (__bf16)0.0f;
}

__global__ void k_zero(float* __restrict__ p, size_t n) {
  size_t i = (size_t)blockIdx.x * blockDim.x + threadIdx.x;
  if (i < n) p[i] = 0.0f;
}

// ---------------------------------------------------------------------------
// WMMA GEMM: C[slab][M][64] = A[M][Kp](bf16) @ Bswz[slab] (+bias on bias_slab).
// One wave per 16-row strip. A-tile: two 16B loads/lane; B-tile: one 32B
// load/lane from the swizzled layout. 4 WMMAs per 32-K step.
__global__ void k_gemm_bf16(const __bf16* __restrict__ A, int Kp,
                            const __bf16* __restrict__ Bswz,
                            const float* __restrict__ bias, int bias_slab,
                            float* __restrict__ C, int M) {
  int row0 = blockIdx.x * 16;
  int slab = blockIdx.y;
  int lane = threadIdx.x & 31;
  int ktiles = Kp >> 5;

  const __bf16* Ar = A + (size_t)(row0 + (lane & 15)) * Kp + ((lane >> 4) << 3);
  const __bf16* Bl = Bswz + (size_t)slab * ktiles * 2048 + lane * 16;
  float* Cs = C + (size_t)slab * (size_t)M * 64;

  v8f acc0 = {}, acc1 = {}, acc2 = {}, acc3 = {};
  for (int kt = 0; kt < ktiles; ++kt) {
    v8bf lo = *(const v8bf*)(Ar);
    v8bf hi = *(const v8bf*)(Ar + 16);
    v16bf a = __builtin_shufflevector(lo, hi, 0, 1, 2, 3, 4, 5, 6, 7,
                                      8, 9, 10, 11, 12, 13, 14, 15);
    v16bf b0 = *(const v16bf*)(Bl + 0 * 512);
    v16bf b1 = *(const v16bf*)(Bl + 1 * 512);
    v16bf b2 = *(const v16bf*)(Bl + 2 * 512);
    v16bf b3 = *(const v16bf*)(Bl + 3 * 512);
    acc0 = __builtin_amdgcn_wmma_f32_16x16x32_bf16(false, a, false, b0, (short)0, acc0, false, false);
    acc1 = __builtin_amdgcn_wmma_f32_16x16x32_bf16(false, a, false, b1, (short)0, acc1, false, false);
    acc2 = __builtin_amdgcn_wmma_f32_16x16x32_bf16(false, a, false, b2, (short)0, acc2, false, false);
    acc3 = __builtin_amdgcn_wmma_f32_16x16x32_bf16(false, a, false, b3, (short)0, acc3, false, false);
    Ar += 32;
    Bl += 2048;
  }
  // C/D layout: lane n=lane&15 -> column; rows mo..mo+7 live in acc[0..7]
  int n = lane & 15;
  int mo = (lane >> 4) * 8;
  bool use_bias = (bias != nullptr) && (slab == bias_slab);
  v8f* accs[4] = {&acc0, &acc1, &acc2, &acc3};
#pragma unroll
  for (int t = 0; t < 4; ++t) {
    int h = t * 16 + n;
    float bv = use_bias ? bias[h] : 0.0f;
#pragma unroll
    for (int r = 0; r < 8; ++r)
      Cs[(size_t)(row0 + mo + r) * 64 + h] = (*accs[t])[r] + bv;
  }
}

// ---- edge scatter: sums[(r*N+dst)*64+h] += xr[(r*N+src)*64+h]; cnt += 1 ----
__global__ void k_aggregate(const float* __restrict__ xr,
                            const int* __restrict__ row,
                            const int* __restrict__ col,
                            const int* __restrict__ etype,
                            float* __restrict__ sums,
                            float* __restrict__ cnt) {
  size_t gid = (size_t)blockIdx.x * blockDim.x + threadIdx.x;
  size_t e = gid >> 4;           // 16 threads cooperate on one edge (64 floats)
  int t = (int)(gid & 15);
  if (e >= (size_t)E_EDGES) return;
  int r = etype[e], s = row[e], d = col[e];
  const float* src = xr   + (((size_t)r * N_NODES + s) << 6);
  float*       dst = sums + (((size_t)r * N_NODES + d) << 6);
  __builtin_prefetch(src, 0, 1);           // global_prefetch_b8 hint
  int h = t << 2;
  float4 v = *reinterpret_cast<const float4*>(src + h);
  atomicAdd(dst + h + 0, v.x);
  atomicAdd(dst + h + 1, v.y);
  atomicAdd(dst + h + 2, v.z);
  atomicAdd(dst + h + 3, v.w);
  if (t == 0) atomicAdd(cnt + (size_t)r * N_NODES + d, 1.0f);
}

// ---- finalize: v = sum_r sums/max(cnt,1) + xroot; act; write bf16 (+f32) ---
// mode 0: ReLU; mode 1: LayerNorm(ln_g, ln_b). block = (64, 4).
__global__ void k_finalize(const float* __restrict__ sums,
                           const float* __restrict__ cnt,
                           const float* __restrict__ xroot,
                           const float* __restrict__ ln_g,
                           const float* __restrict__ ln_b,
                           float* __restrict__ outf,
                           __bf16* __restrict__ outb, int mode) {
  __shared__ float red[4][64];
  int node = blockIdx.x * 4 + threadIdx.y;    // 50000/4 = 12500 blocks exact
  int h = threadIdx.x;
  float v = xroot[(size_t)node * 64 + h];
#pragma unroll
  for (int r = 0; r < N_REL; ++r) {
    size_t idx = (size_t)r * N_NODES + node;
    v += sums[idx * 64 + h] / fmaxf(cnt[idx], 1.0f);
  }
  if (mode == 0) {
    v = fmaxf(v, 0.0f);
  } else {
    // LayerNorm over H=64 (block spans 2 waves per row -> LDS tree reduce)
    red[threadIdx.y][h] = v;
    __syncthreads();
#pragma unroll
    for (int off = 32; off > 0; off >>= 1) {
      if (h < off) red[threadIdx.y][h] += red[threadIdx.y][h + off];
      __syncthreads();
    }
    float mu = red[threadIdx.y][0] * (1.0f / 64.0f);
    __syncthreads();
    float dv = v - mu;
    red[threadIdx.y][h] = dv * dv;
    __syncthreads();
#pragma unroll
    for (int off = 32; off > 0; off >>= 1) {
      if (h < off) red[threadIdx.y][h] += red[threadIdx.y][h + off];
      __syncthreads();
    }
    float var = red[threadIdx.y][0] * (1.0f / 64.0f);
    v = dv * rsqrtf(var + 1e-5f) * ln_g[h] + ln_b[h];
  }
  outb[(size_t)node * 64 + h] = (__bf16)v;
  if (outf) outf[(size_t)node * 64 + h] = v;
}

// ---------------------------------------------------------------------------
// Fused edge classifier: 16 edges per wave.
// ef = [x3[row], x3[col]] (16x128 bf16) @ ew1 (128x64) via 16 WMMAs,
// then +eb1, ReLU, @ew2 (64x3) in-register, shfl-xor reduce, log_softmax.
__global__ void k_edge_mlp(const __bf16* __restrict__ x3b,
                           const int* __restrict__ row,
                           const int* __restrict__ col,
                           const __bf16* __restrict__ ew1swz,  // [4kt][4nt][32][16]
                           const float* __restrict__ eb1,
                           const float* __restrict__ ew2,
                           const float* __restrict__ eb2,
                           float* __restrict__ edge_out) {
  int wave = blockIdx.x * (blockDim.x >> 5) + (threadIdx.x >> 5);
  size_t e0 = (size_t)wave * 16;           // E = 1.6M divisible by 16
  if (e0 >= (size_t)E_EDGES) return;
  int lane = threadIdx.x & 31;
  int m = lane & 15;
  int kb8 = (lane >> 4) << 3;
  size_t em = e0 + m;
  const __bf16* prb = x3b + (size_t)row[em] * 64 + kb8;
  const __bf16* pcb = x3b + (size_t)col[em] * 64 + kb8;
  __builtin_prefetch(prb, 0, 1);
  __builtin_prefetch(pcb, 0, 1);
  const __bf16* Bl = ew1swz + lane * 16;

  v8f acc0 = {}, acc1 = {}, acc2 = {}, acc3 = {};
#pragma unroll
  for (int kc = 0; kc < 4; ++kc) {
    const __bf16* pb = (kc < 2) ? prb : pcb;   // static concat structure
    int koff = (kc & 1) * 32;
    v8bf lo = *(const v8bf*)(pb + koff);
    v8bf hi = *(const v8bf*)(pb + koff + 16);
    v16bf a = __builtin_shufflevector(lo, hi, 0, 1, 2, 3, 4, 5, 6, 7,
                                      8, 9, 10, 11, 12, 13, 14, 15);
    const __bf16* Bt = Bl + kc * 2048;
    v16bf b0 = *(const v16bf*)(Bt + 0 * 512);
    v16bf b1 = *(const v16bf*)(Bt + 1 * 512);
    v16bf b2 = *(const v16bf*)(Bt + 2 * 512);
    v16bf b3 = *(const v16bf*)(Bt + 3 * 512);
    acc0 = __builtin_amdgcn_wmma_f32_16x16x32_bf16(false, a, false, b0, (short)0, acc0, false, false);
    acc1 = __builtin_amdgcn_wmma_f32_16x16x32_bf16(false, a, false, b1, (short)0, acc1, false, false);
    acc2 = __builtin_amdgcn_wmma_f32_16x16x32_bf16(false, a, false, b2, (short)0, acc2, false, false);
    acc3 = __builtin_amdgcn_wmma_f32_16x16x32_bf16(false, a, false, b3, (short)0, acc3, false, false);
  }

  int n = lane & 15;
  int mo = (lane >> 4) * 8;
  float p[8][3];
#pragma unroll
  for (int r2 = 0; r2 < 8; ++r2) { p[r2][0] = p[r2][1] = p[r2][2] = 0.0f; }
  v8f* accs[4] = {&acc0, &acc1, &acc2, &acc3};
#pragma unroll
  for (int t = 0; t < 4; ++t) {
    int h = t * 16 + n;
    float b = eb1[h];
    float w0 = ew2[h * 3 + 0], w1 = ew2[h * 3 + 1], w2 = ew2[h * 3 + 2];
#pragma unroll
    for (int r2 = 0; r2 < 8; ++r2) {
      float eh = fmaxf((*accs[t])[r2] + b, 0.0f);
      p[r2][0] += eh * w0;
      p[r2][1] += eh * w1;
      p[r2][2] += eh * w2;
    }
  }
  // reduce the 16 h-columns held across each 16-lane group
#pragma unroll
  for (int off = 8; off >= 1; off >>= 1) {
#pragma unroll
    for (int r2 = 0; r2 < 8; ++r2) {
      p[r2][0] += __shfl_xor(p[r2][0], off, 16);
      p[r2][1] += __shfl_xor(p[r2][1], off, 16);
      p[r2][2] += __shfl_xor(p[r2][2], off, 16);
    }
  }
  if (n < 8) {
    int r2 = n;
    size_t e = e0 + mo + r2;
    float v0 = p[r2][0] + eb2[0];
    float v1 = p[r2][1] + eb2[1];
    float v2 = p[r2][2] + eb2[2];
    float mx = fmaxf(v0, fmaxf(v1, v2));
    float lse = mx + logf(expf(v0 - mx) + expf(v1 - mx) + expf(v2 - mx));
    edge_out[e * 3 + 0] = v0 - lse;
    edge_out[e * 3 + 1] = v1 - lse;
    edge_out[e * 3 + 2] = v2 - lse;
  }
}

// ---- node classifier (tiny: 0.1 G MACs, plain VALU) ------------------------
__global__ void k_node_mlp(const float* __restrict__ x3,
                           const float* __restrict__ nw1,
                           const float* __restrict__ nb1,
                           const float* __restrict__ nw2,
                           const float* __restrict__ nb2,
                           float* __restrict__ node_out) {
  int nidx = blockIdx.x * blockDim.x + threadIdx.x;
  if (nidx >= N_NODES) return;
  const float* xp = x3 + (size_t)nidx * 64;
  float xv[64];
#pragma unroll
  for (int h = 0; h < 64; ++h) xv[h] = xp[h];
  float v0 = nb2[0], v1 = nb2[1];
  for (int j = 0; j < 32; ++j) {
    float hs = nb1[j];
#pragma unroll 8
    for (int h = 0; h < 64; ++h) hs += xv[h] * nw1[h * 32 + j];
    hs = fmaxf(hs, 0.0f);
    v0 += hs * nw2[j * 2 + 0];
    v1 += hs * nw2[j * 2 + 1];
  }
  float mx = fmaxf(v0, v1);
  float lse = mx + logf(expf(v0 - mx) + expf(v1 - mx));
  node_out[(size_t)nidx * 2 + 0] = v0 - lse;
  node_out[(size_t)nidx * 2 + 1] = v1 - lse;
}

// ---------------------------------------------------------------------------
extern "C" void kernel_launch(void* const* d_in, const int* in_sizes, int n_in,
                              void* d_out, int out_size, void* d_ws, size_t ws_size,
                              hipStream_t stream) {
  const float* x      = (const float*)d_in[0];
  const int*   row    = (const int*)d_in[1];
  const int*   col    = row + E_EDGES;
  const int*   etype  = (const int*)d_in[2];
  const float* W1     = (const float*)d_in[3];
  const float* root1  = (const float*)d_in[4];
  const float* b1     = (const float*)d_in[5];
  const float* W2     = (const float*)d_in[6];
  const float* root2  = (const float*)d_in[7];
  const float* b2     = (const float*)d_in[8];
  const float* W3     = (const float*)d_in[9];
  const float* root3  = (const float*)d_in[10];
  const float* b3     = (const float*)d_in[11];
  const float* ln_g   = (const float*)d_in[12];
  const float* ln_b   = (const float*)d_in[13];
  const float* ew1    = (const float*)d_in[14];
  const float* eb1    = (const float*)d_in[15];
  const float* ew2    = (const float*)d_in[16];
  const float* eb2    = (const float*)d_in[17];
  const float* nw1    = (const float*)d_in[18];
  const float* nb1    = (const float*)d_in[19];
  const float* nw2    = (const float*)d_in[20];
  const float* nb2    = (const float*)d_in[21];

  float* edge_out = (float*)d_out;
  float* node_out = (float*)d_out + (size_t)E_EDGES * 3;

  // workspace arena
  char* ws = (char*)d_ws;
  size_t off = 0;
  auto arena = [&](size_t bytes) -> void* {
    void* p = ws + off;
    off += (bytes + 255) & ~(size_t)255;
    return p;
  };
  const int KP1 = 800;                         // 770 padded to 32
  const int KT1 = KP1 / 32, KT2 = 2;           // layer2/3: Kp=64 -> 2 tiles
  __bf16* wswz1 = (__bf16*)arena(sizeof(__bf16) * 4ull * KT1 * 2048);
  __bf16* wswz2 = (__bf16*)arena(sizeof(__bf16) * 4ull * KT2 * 2048);
  __bf16* wswz3 = (__bf16*)arena(sizeof(__bf16) * 4ull * KT2 * 2048);
  __bf16* e1swz = (__bf16*)arena(sizeof(__bf16) * 4ull * 2048);      // K=128
  __bf16* xbf   = (__bf16*)arena(sizeof(__bf16) * (size_t)N_NODES * KP1);
  __bf16* x1bf  = (__bf16*)arena(sizeof(__bf16) * (size_t)N_NODES * 64);
  __bf16* x2bf  = (__bf16*)arena(sizeof(__bf16) * (size_t)N_NODES * 64);
  __bf16* x3bf  = (__bf16*)arena(sizeof(__bf16) * (size_t)N_NODES * 64);
  float*  xr    = (float*)arena(sizeof(float) * 4ull * N_NODES * 64); // r0..r2, root
  float*  sums  = (float*)arena(sizeof(float) * 3ull * N_NODES * 64);
  float*  cnt   = (float*)arena(sizeof(float) * 3ull * N_NODES);
  float*  x3    = (float*)arena(sizeof(float) * (size_t)N_NODES * 64);
  (void)ws_size; (void)n_in; (void)in_sizes; (void)out_size;

  auto packw = [&](const float* src, __bf16* dst, int K, int Kp, int nslab) {
    size_t n = (size_t)nslab * (Kp >> 5) * 2048;
    k_pack_w<<<(unsigned)((n + 255) / 256), 256, 0, stream>>>(src, dst, K, Kp, nslab);
  };
  // weights: W[r] slabs 0..2 then root as slab 3
  packw(W1, wswz1, 770, KP1, 3);
  packw(root1, wswz1 + 3ull * KT1 * 2048, 770, KP1, 1);
  packw(W2, wswz2, 64, 64, 3);
  packw(root2, wswz2 + 3ull * KT2 * 2048, 64, 64, 1);
  packw(W3, wswz3, 64, 64, 3);
  packw(root3, wswz3 + 3ull * KT2 * 2048, 64, 64, 1);
  packw(ew1, e1swz, 128, 128, 1);

  // activations: x -> padded bf16
  {
    size_t n = (size_t)N_NODES * KP1;
    k_pack_act<<<(unsigned)((n + 255) / 256), 256, 0, stream>>>(x, xbf, 770, KP1, N_NODES);
  }

  const size_t sums_n = 3ull * N_NODES * 64;
  const size_t cnt_n  = 3ull * N_NODES;
  const float* xroot = xr + 3ull * N_NODES * 64;
  dim3 fin_blk(64, 4);
  unsigned agg_blocks = (unsigned)(((size_t)E_EDGES * 16) / 256);

  // ---- layer 1: x [N,770] -> x1 (bf16)
  k_gemm_bf16<<<dim3(N_NODES / 16, 4), 32, 0, stream>>>(xbf, KP1, wswz1, b1, 3, xr, N_NODES);
  k_zero<<<(unsigned)((sums_n + 255) / 256), 256, 0, stream>>>(sums, sums_n);
  k_zero<<<(unsigned)((cnt_n + 255) / 256), 256, 0, stream>>>(cnt, cnt_n);
  k_aggregate<<<agg_blocks, 256, 0, stream>>>(xr, row, col, etype, sums, cnt);
  k_finalize<<<N_NODES / 4, fin_blk, 0, stream>>>(sums, cnt, xroot, ln_g, ln_b,
                                                  nullptr, x1bf, 0);

  // ---- layer 2: x1 -> x2 (bf16)
  k_gemm_bf16<<<dim3(N_NODES / 16, 4), 32, 0, stream>>>(x1bf, 64, wswz2, b2, 3, xr, N_NODES);
  k_zero<<<(unsigned)((sums_n + 255) / 256), 256, 0, stream>>>(sums, sums_n);
  k_zero<<<(unsigned)((cnt_n + 255) / 256), 256, 0, stream>>>(cnt, cnt_n);
  k_aggregate<<<agg_blocks, 256, 0, stream>>>(xr, row, col, etype, sums, cnt);
  k_finalize<<<N_NODES / 4, fin_blk, 0, stream>>>(sums, cnt, xroot, ln_g, ln_b,
                                                  nullptr, x2bf, 0);

  // ---- layer 3: x2 -> x3 (LayerNorm; f32 for node head + bf16 for edge head)
  k_gemm_bf16<<<dim3(N_NODES / 16, 4), 32, 0, stream>>>(x2bf, 64, wswz3, b3, 3, xr, N_NODES);
  k_zero<<<(unsigned)((sums_n + 255) / 256), 256, 0, stream>>>(sums, sums_n);
  k_zero<<<(unsigned)((cnt_n + 255) / 256), 256, 0, stream>>>(cnt, cnt_n);
  k_aggregate<<<agg_blocks, 256, 0, stream>>>(xr, row, col, etype, sums, cnt);
  k_finalize<<<N_NODES / 4, fin_blk, 0, stream>>>(sums, cnt, xroot, ln_g, ln_b,
                                                  x3, x3bf, 1);

  // ---- heads
  k_edge_mlp<<<(E_EDGES / 16 + 3) / 4, 128, 0, stream>>>(
      x3bf, row, col, e1swz, eb1, ew2, eb2, edge_out);
  k_node_mlp<<<(N_NODES + 255) / 256, 256, 0, stream>>>(
      x3, nw1, nb1, nw2, nb2, node_out);
}
